// ScaledDotProductAttention_Operator_80547816669370
// MI455X (gfx1250) — matrix operand order repaired and checked by reading, not compiled, not run on hardware
//
#include <hip/hip_runtime.h>
#include <hip/hip_bf16.h>

// ---------------------------------------------------------------------------
// Shapes (from reference): B=4, nH=4, T=64, H=32, W=32, d=8
//   BN  = B*nH = 16          (independent attention batches)
//   T   = 64                 (tokens per batch)
//   F   = H*W*d = 8192       (flattened feature dim per token)
// kx  = (Q Kᵀ)/1024 : 16 x (64x64), K-dim 8192       -> bf16 WMMA kernel
// out = conv_gauss( kx @ V ) fused per output token  -> LDS kernel
// ---------------------------------------------------------------------------

typedef __attribute__((ext_vector_type(16))) __bf16 v16bf;
typedef __attribute__((ext_vector_type(8)))  float  v8f;

#define BN 16
#define TT 64
#define FF 8192   // 32*32*8

// ===========================================================================
// Kernel 1: kx[bn, p, q] = (1/1024) * sum_k Q[bn,p,k] * K[bn,q,k]
//   grid  = 64 blocks  (bn * 4 row-tiles)
//   block = 128 threads (4 waves); each wave -> one 16x16 tile (col tile = wave)
//   K loop: 8192 in steps of 32, unrolled x2 -> v_wmma_f32_16x16x32_bf16
// ===========================================================================
__global__ __launch_bounds__(128)
void qkT_wmma_kernel(const float* __restrict__ Q,
                     const float* __restrict__ K,
                     float* __restrict__ KX)
{
    const int bn   = blockIdx.x >> 2;          // 0..15
    const int mi   = (blockIdx.x & 3) * 16;    // row-tile base within 64
    const int ni   = (threadIdx.x >> 5) * 16;  // col-tile base (wave id)
    const int lane = threadIdx.x & 31;
    const int l16  = lane & 15;
    const bool hi  = (lane & 16) != 0;

    // A fragment rows come from Q[p=mi+l16], B fragment cols from K[q=ni+l16]
    const float* qrow = Q + ((size_t)bn * TT + (mi + l16)) * FF;
    const float* krow = K + ((size_t)bn * TT + (ni + l16)) * FF;

    // 16-bit A 16x32 layout: lanes 0-15 hold K {0..7,16..23}, lanes 16-31 hold
    // K {8..15,24..31}.  16-bit B 32x16: same K order as A (N = lane&15).
    const int ak = hi ? 8  : 0;
    const int bk = hi ? 8  : 0;

    v8f c = {};
    for (int kk = 0; kk < FF; kk += 64) {
        // prefetch next K-slice of both operand rows into cache
        __builtin_prefetch(qrow + kk + 64, 0, 3);
        __builtin_prefetch(krow + kk + 64, 0, 3);
#pragma unroll
        for (int u = 0; u < 2; ++u) {
            const int kb = kk + u * 32;
            const float* qa = qrow + kb + ak;
            const float* ka = krow + kb + bk;

            float4 a0 = ((const float4*)(qa     ))[0];
            float4 a1 = ((const float4*)(qa     ))[1];
            float4 a2 = ((const float4*)(qa + 16))[0];
            float4 a3 = ((const float4*)(qa + 16))[1];
            float4 b0 = ((const float4*)ka)[0];
            float4 b1 = ((const float4*)ka)[1];
            float4 b2 = ((const float4*)(ka + 16))[0];
            float4 b3 = ((const float4*)(ka + 16))[1];

            float af[16] = {a0.x,a0.y,a0.z,a0.w, a1.x,a1.y,a1.z,a1.w,
                            a2.x,a2.y,a2.z,a2.w, a3.x,a3.y,a3.z,a3.w};
            float bf[16] = {b0.x,b0.y,b0.z,b0.w, b1.x,b1.y,b1.z,b1.w,
                            b2.x,b2.y,b2.z,b2.w, b3.x,b3.y,b3.z,b3.w};
            v16bf A, B;
#pragma unroll
            for (int t = 0; t < 16; ++t) {
                A[t] = (__bf16)af[t];
                B[t] = (__bf16)bf[t];
            }
            c = __builtin_amdgcn_wmma_f32_16x16x32_bf16(
                    false, A, false, B, (short)0, c, false, false);
        }
    }

    // C/D f32 16x16 layout: VGPR r -> (M = (hi?8:0)+r, N = lane&15)
    const int m = mi + (hi ? 8 : 0);
    const int n = ni + l16;
    float* o = KX + bn * (TT * TT) + n;
#pragma unroll
    for (int r = 0; r < 8; ++r)
        o[(m + r) * TT] = c[r] * (1.0f / 1024.0f);
}

// ===========================================================================
// Kernel 2: out[bn,p,:,:,:] = gauss_circ_conv( sum_q kx[bn,p,q] * V[bn,q,:,:,:] )
//   grid  = 1024 blocks (bn*64 + p), block = 256 threads
//   Y (32KB) lives in LDS; conv done as two separable 32-tap circular passes
//   with weights u[j] = column-sum of the (normalized) 2-D input kernel,
//   which is exact because the Gaussian kernel is an outer product.
// ===========================================================================
__global__ __launch_bounds__(256)
void mix_conv_kernel(const float* __restrict__ V,
                     const float* __restrict__ KER,   // (1,1,32,32), sums to 1
                     const float* __restrict__ KX,
                     float* __restrict__ OUT)
{
    __shared__ float sKx[TT];
    __shared__ float sG[32];
    __shared__ float sY[FF];   // pre-conv token image (h,w,d)
    __shared__ float sT[FF];   // after W-pass

    const int bn  = blockIdx.x >> 6;
    const int p   = blockIdx.x & 63;
    const int tid = threadIdx.x;

    if (tid < TT) sKx[tid] = KX[bn * (TT * TT) + p * TT + tid];
    if (tid < 32) {
        float s = 0.0f;
        for (int i = 0; i < 32; ++i) s += KER[i * 32 + tid];  // u[j]
        sG[tid] = s;
    }
    __syncthreads();

    // ---- stage 1: Y = sum_q kx[p,q] * V[bn,q,:]  (registers, float4) -------
    float4 acc[8];
#pragma unroll
    for (int i = 0; i < 8; ++i) acc[i] = make_float4(0.f, 0.f, 0.f, 0.f);

    const float4* vb = (const float4*)(V + (size_t)bn * TT * FF);
    for (int q = 0; q < TT; ++q) {
        const float  w  = sKx[q];
        const float4* vr = vb + (size_t)q * (FF / 4);
        __builtin_prefetch(vr + (FF / 4), 0, 1);   // next token's V row
#pragma unroll
        for (int i = 0; i < 8; ++i) {
            float4 x = vr[tid + i * 256];
            acc[i].x = fmaf(w, x.x, acc[i].x);
            acc[i].y = fmaf(w, x.y, acc[i].y);
            acc[i].z = fmaf(w, x.z, acc[i].z);
            acc[i].w = fmaf(w, x.w, acc[i].w);
        }
    }
    float4* sY4 = (float4*)sY;
#pragma unroll
    for (int i = 0; i < 8; ++i) sY4[tid + i * 256] = acc[i];
    __syncthreads();

    // ---- stage 2: circular conv along W:  t[h,w,d] = sum_j u[j] Y[h,(w+j-16)%32,d]
#pragma unroll 4
    for (int i = 0; i < 32; ++i) {
        const int e = tid + i * 256;
        const int d = e & 7, w = (e >> 3) & 31, h = e >> 8;
        float s = 0.0f;
#pragma unroll
        for (int j = 0; j < 32; ++j)
            s = fmaf(sG[j], sY[(((h << 5) + ((w + j + 16) & 31)) << 3) + d], s);
        sT[e] = s;
    }
    __syncthreads();

    // ---- stage 3: circular conv along H, write out -------------------------
    float* ob = OUT + ((size_t)bn * TT + p) * FF;
#pragma unroll 4
    for (int i = 0; i < 32; ++i) {
        const int e = tid + i * 256;
        const int d = e & 7, w = (e >> 3) & 31, h = e >> 8;
        float s = 0.0f;
#pragma unroll
        for (int j = 0; j < 32; ++j)
            s = fmaf(sG[j], sT[(((((h + j + 16) & 31) << 5) + w) << 3) + d], s);
        ob[e] = s;
    }
}

// ===========================================================================
extern "C" void kernel_launch(void* const* d_in, const int* in_sizes, int n_in,
                              void* d_out, int out_size, void* d_ws, size_t ws_size,
                              hipStream_t stream)
{
    const float* q   = (const float*)d_in[0];  // (4,4,64,32,32,8) f32
    const float* k   = (const float*)d_in[1];
    const float* v   = (const float*)d_in[2];
    const float* ker = (const float*)d_in[3];  // (1,1,32,32) f32
    float* out = (float*)d_out;
    float* kx  = (float*)d_ws;                 // needs 16*64*64*4 = 256 KB

    qkT_wmma_kernel<<<BN * 4, 128, 0, stream>>>(q, k, kx);
    mix_conv_kernel<<<BN * TT, 256, 0, stream>>>(v, ker, kx, out);
}
